// ParticleNet_57801669869645
// MI455X (gfx1250) — matrix operand order repaired and checked by reading, not compiled, not run on hardware
//
#include <hip/hip_runtime.h>
#include <hip/hip_bf16.h>

// ---------------------------------------------------------------------------
// ParticleNet (DGCNN) forward for MI455X / gfx1250.
// Heavy math (1x1 edge convs, ~267 GFLOP/call) runs on v_wmma_f32_16x16x32_f16
// (f16 in, f32 accumulate). Activations are LDS-resident per 16-point chunk;
// the fts table is staged into LDS with GLOBAL_LOAD_ASYNC_TO_LDS_B128 from an
// f16 [b][n][C] copy produced by the previous stage (ASYNCcnt-tracked).
//
// d_in layout assumption (top level in setup_inputs() order, params flattened
// with sorted dict keys, jax tree-flatten style):
//   0: points (256,2,128)  1: features (256,7,128)  2: global_features (256,4,1)
//   3: bn_fts g (7)        4: bn_fts b (7)
//   block i (i=0..2), base Ei = 5 + 12*i:
//     Ei+0..5 : bn g0,b0,g1,b1,g2,b2   Ei+6..8 : conv W0,W1,W2
//     Ei+9    : sc W                   Ei+10,11: sc_bn g,b
//   41: fc W (256,256)   42: fc b (256)
//   43: comb W1 (128,320) 44: comb b1 (128) 45: comb W2 (2,128) 46: comb b2 (2)
//   47: glob W (64,4)     48: glob b (64)
// ---------------------------------------------------------------------------

#define B_   256
#define N_   128
#define K_   16
#define EPS_ 1e-5f

typedef _Float16 v8h  __attribute__((ext_vector_type(8)));
typedef _Float16 v16h __attribute__((ext_vector_type(16)));
typedef float    v8f  __attribute__((ext_vector_type(8)));

// ---------------------------------------------------------------------------
// WMMA fragment loaders (wave32 layouts from CDNA5 ISA §7.12.2)
// A: 16x32 f16, lane L: m = L&15; halves 0..7 -> K = (L>>4)*8 + 0..7,
//    halves 8..15 -> K = (L>>4)*8 + 16..23   (two 16B loads per lane)
// B: 32x16 f16, lane L: n = L&15; same K layout as A (two 16B loads per lane)
// C/D: 8 f32, lane L: n = L&15; vgpr r -> m = (L>>4)*8 + r
// ---------------------------------------------------------------------------
__device__ __forceinline__ v16h load_frag_a(const _Float16* src, int stride,
                                            int rowBase, int kBase, int lane) {
  const int m  = lane & 15;
  const int hg = lane >> 4;
  const _Float16* p = src + (size_t)(rowBase + m) * stride + kBase + hg * 8;
  v16h a;
  *(v8h*)&a       = *(const v8h*)(p);
  *((v8h*)&a + 1) = *(const v8h*)(p + 16);
  return a;
}

__device__ __forceinline__ v16h load_frag_b(const _Float16* Wt, int stride,
                                            int o0, int kBase, int lane) {
  const int n  = lane & 15;
  const int hg = lane >> 4;
  const _Float16* p = Wt + (size_t)(o0 + n) * stride + kBase + hg * 8;
  v16h b;
  *(v8h*)&b       = *(const v8h*)(p);
  *((v8h*)&b + 1) = *(const v8h*)(p + 16);
  return b;
}

// ---------------------------------------------------------------------------
// prep: mask, counts, batchnorm of input features -> fts0H (f16, [b][n][32])
// ---------------------------------------------------------------------------
__global__ void prep_kernel(const float* __restrict__ features,
                            const float* __restrict__ bn_g,
                            const float* __restrict__ bn_b,
                            float* __restrict__ mask,
                            float* __restrict__ counts,
                            _Float16* __restrict__ fts0H) {
  __shared__ float red[N_];
  const int b = blockIdx.x;
  const int n = threadIdx.x;
  float s = 0.f;
  for (int c = 0; c < 7; ++c)
    s += fabsf(features[((size_t)b * 7 + c) * N_ + n]);
  const float m = (s != 0.f) ? 1.f : 0.f;
  mask[b * N_ + n] = m;
  red[n] = m;
  __syncthreads();
  for (int st = N_ / 2; st > 0; st >>= 1) {
    if (n < st) red[n] += red[n + st];
    __syncthreads();
  }
  if (n == 0) counts[b] = fmaxf(red[0], 1.f);
  const float bnscale = rsqrtf(1.f + EPS_);
  _Float16* dst = fts0H + ((size_t)b * N_ + n) * 32;
  for (int c = 0; c < 32; ++c) {
    float v = 0.f;
    if (c < 7) {
      const float f = features[((size_t)b * 7 + c) * N_ + n];
      v = m * (f * bn_g[c] * bnscale + bn_b[c]);
    }
    dst[c] = (_Float16)v;
  }
}

// ---------------------------------------------------------------------------
// Weight repack f32 -> f16 (with channel padding)
// ---------------------------------------------------------------------------
__global__ void conv_w_edge0(const float* __restrict__ W, _Float16* __restrict__ dst,
                             int C_in, int Cpad, int cout) {
  const int i = blockIdx.x * blockDim.x + threadIdx.x;
  const int tot = cout * 2 * Cpad;
  if (i >= tot) return;
  const int o = i / (2 * Cpad);
  const int col = i % (2 * Cpad);
  float v = 0.f;
  if (col < Cpad) {
    if (col < C_in) v = W[o * (2 * C_in) + col];            // center channels
  } else {
    const int c = col - Cpad;
    if (c < C_in) v = W[o * (2 * C_in) + C_in + c];         // (nbr - ctr) channels
  }
  dst[i] = (_Float16)v;
}

__global__ void conv_w_plain(const float* __restrict__ W, _Float16* __restrict__ dst, int n) {
  const int i = blockIdx.x * blockDim.x + threadIdx.x;
  if (i < n) dst[i] = (_Float16)W[i];
}

__global__ void conv_w_sc(const float* __restrict__ W, _Float16* __restrict__ dst,
                          int C_in, int Cpad, int cout) {
  const int i = blockIdx.x * blockDim.x + threadIdx.x;
  if (i >= cout * Cpad) return;
  const int o = i / Cpad;
  const int c = i % Cpad;
  dst[i] = (_Float16)((c < C_in) ? W[o * C_in + c] : 0.f);
}

// ---------------------------------------------------------------------------
// knn: per-batch block, per-point thread; shifted coords in LDS; top-16 via
// insertion sort (reference takes top-17 of -dist^2 and drops self).
// ---------------------------------------------------------------------------
__global__ void knn_kernel(const float* __restrict__ coords,
                           const float* __restrict__ mask,
                           int C, int* __restrict__ idxOut) {
  extern __shared__ __align__(16) char smK[];
  float* cs = (float*)smK;           // [C][128] shifted coords
  float* mk = cs + C * N_;           // [128]
  const int b = blockIdx.x;
  const int t = threadIdx.x;
  mk[t] = mask[b * N_ + t];
  __syncthreads();
  for (int i = t; i < C * N_; i += N_) {
    const int c = i >> 7;
    const int m = i & (N_ - 1);
    const float mv = mk[m];
    cs[i] = coords[((size_t)b * C + c) * N_ + m] * mv + (1.f - mv) * 1e9f;
  }
  __syncthreads();
  float bd[K_]; int bi[K_];
#pragma unroll
  for (int k = 0; k < K_; ++k) { bd[k] = 3.0e38f; bi[k] = 0; }
  const int n = t;
  for (int m = 0; m < N_; ++m) {
    if (m == n) continue;
    float d = 0.f;
    for (int c = 0; c < C; ++c) {
      const float df = cs[c * N_ + n] - cs[c * N_ + m];
      d = fmaf(df, df, d);
    }
    if (d < bd[K_ - 1]) {
      int j = K_ - 1;
      while (j > 0 && bd[j - 1] > d) { bd[j] = bd[j - 1]; bi[j] = bi[j - 1]; --j; }
      bd[j] = d; bi[j] = m;
    }
  }
  for (int k = 0; k < K_; ++k) idxOut[((size_t)b * N_ + n) * K_ + k] = bi[k];
}

// ---------------------------------------------------------------------------
// One GEMM layer: dst[256][Wd] = relu(bn(src[256][Wd] x Wt^T)), all in LDS,
// 8 waves cover 16 x (Wd/16) 16x16 output tiles, K stepped by 32 via WMMA.
// ---------------------------------------------------------------------------
__device__ __forceinline__ void gemm_bn_relu(const _Float16* src, _Float16* dst,
                                             const _Float16* __restrict__ Wt,
                                             const float* __restrict__ gg,
                                             const float* __restrict__ bb,
                                             int Wd, int lane, int wave) {
  const float bnscale = rsqrtf(1.f + EPS_);
  const int totalTiles = 16 * (Wd >> 4);      // 64 / 128 / 256 (multiple of 8)
  for (int tile = wave; tile < totalTiles; tile += 8) {
    const int mt = tile & 15;
    const int nt = tile >> 4;
    const int rowBase = mt * 16;
    const int o0 = nt * 16;
    v8f acc = {0.f, 0.f, 0.f, 0.f, 0.f, 0.f, 0.f, 0.f};
    for (int kk = 0; kk < Wd; kk += 32) {
      const v16h a = load_frag_a(src, Wd, rowBase, kk, lane);
      const v16h bf = load_frag_b(Wt, Wd, o0, kk, lane);
      acc = __builtin_amdgcn_wmma_f32_16x16x32_f16(false, a, false, bf,
                                                   (short)0, acc, false, false);
    }
    const int o = o0 + (lane & 15);
    const float s = gg[o] * bnscale;
    const float tsh = bb[o];
    const int rbase = rowBase + ((lane >> 4) << 3);
#pragma unroll
    for (int r = 0; r < 8; ++r) {
      const float v = fmaxf(acc[r] * s + tsh, 0.f);
      dst[(size_t)(rbase + r) * Wd + o] = (_Float16)v;
    }
  }
}

// ---------------------------------------------------------------------------
// Fused EdgeConv block: one workgroup = (batch b, 16 points -> 256 edges).
// LDS: fts table [128][Cpad] f16 + two [256][Wd] f16 activation buffers.
// fts table staged via async global->LDS copies (f16 source is already in the
// exact [n][Cpad] LDS layout, so this is a straight byte copy).
// ---------------------------------------------------------------------------
__global__ void edge_block_kernel(
    const _Float16* __restrict__ ftsInH,          // f16 [b][n][Cpad]
    float* __restrict__ ftsOutF,                  // f32 [b][o][n]   (knn/head)
    _Float16* __restrict__ ftsOutH,               // f16 [b][n][Wd]  (next stage)
    const int* __restrict__ nbrIdx, const float* __restrict__ mask,
    const _Float16* __restrict__ W0, const _Float16* __restrict__ W1,
    const _Float16* __restrict__ W2, const _Float16* __restrict__ Wsc,
    const float* __restrict__ ga0, const float* __restrict__ be0,
    const float* __restrict__ ga1, const float* __restrict__ be1,
    const float* __restrict__ ga2, const float* __restrict__ be2,
    const float* __restrict__ gsc, const float* __restrict__ bsc,
    int Cpad, int Wd) {
  extern __shared__ __align__(16) char sm[];
  _Float16* ftsS = (_Float16*)sm;                   // [128][Cpad]
  _Float16* xb   = ftsS + (size_t)N_ * Cpad;        // [256][Wd]
  _Float16* yb   = xb + (size_t)256 * Wd;           // [256][Wd]

  const int b  = blockIdx.x >> 3;
  const int n0 = (blockIdx.x & 7) * 16;
  const int t  = threadIdx.x;                       // 256 threads = 8 waves
  const int lane = t & 31;
  const int wave = t >> 5;

  // ---- Stage fts table (128 x Cpad f16) into LDS via async copy -----------
  {
    const unsigned long long srcBase =
        (unsigned long long)(uintptr_t)(ftsInH + (size_t)b * N_ * Cpad);
    const unsigned ldsBase = (unsigned)(uintptr_t)ftsS;   // addr[31:0] = LDS offset
    const int totalBytes = N_ * Cpad * 2;                 // 8/16/32 KB
    for (int byteOff = t * 16; byteOff < totalBytes; byteOff += 256 * 16) {
      const unsigned ldsAddr = ldsBase + (unsigned)byteOff;
      const unsigned vOff = (unsigned)byteOff;
      asm volatile("global_load_async_to_lds_b128 %0, %1, %2"
                   :: "v"(ldsAddr), "v"(vOff), "s"(srcBase)
                   : "memory");
    }
    asm volatile("s_wait_asynccnt 0x0" ::: "memory");
  }
  __syncthreads();

  // Build edge features: row e = nl*16 + k; cols [0,Cpad)=ctr, [Cpad,2Cpad)=nbr-ctr.
  {
    const int nl = t >> 4;
    const int k  = t & 15;
    const int n  = n0 + nl;
    const int nb = nbrIdx[((size_t)b * N_ + n) * K_ + k];
    _Float16* dst = xb + (size_t)t * Wd;            // Wd == 2*Cpad
    const _Float16* fc = ftsS + (size_t)n * Cpad;
    const _Float16* fn = ftsS + (size_t)nb * Cpad;
    for (int c = 0; c < Cpad; ++c) {
      const _Float16 cv = fc[c];
      dst[c]        = cv;
      dst[Cpad + c] = (_Float16)(fn[c] - cv);
    }
  }
  __syncthreads();

  gemm_bn_relu(xb, yb, W0, ga0, be0, Wd, lane, wave);
  __syncthreads();
  gemm_bn_relu(yb, xb, W1, ga1, be1, Wd, lane, wave);
  __syncthreads();
  gemm_bn_relu(xb, yb, W2, ga2, be2, Wd, lane, wave);
  __syncthreads();

  // Mean over k + shortcut + relu + mask; write f32 [b][o][n] and f16 [b][n][o].
  const float bnscale = rsqrtf(1.f + EPS_);
  for (int i = t; i < 16 * Wd; i += 256) {
    const int o  = i & (Wd - 1);
    const int nl = i / Wd;
    const int n  = n0 + nl;
    float agg = 0.f;
    const _Float16* yr = yb + (size_t)(nl * 16) * Wd + o;
#pragma unroll
    for (int k = 0; k < K_; ++k) agg += (float)yr[(size_t)k * Wd];
    agg *= (1.f / 16.f);
    float sc = 0.f;
    const _Float16* fr = ftsS + (size_t)n * Cpad;
    const _Float16* wr = Wsc + (size_t)o * Cpad;
    for (int c = 0; c < Cpad; ++c) sc += (float)fr[c] * (float)wr[c];
    sc = sc * (gsc[o] * bnscale) + bsc[o];
    const float mv = mask[b * N_ + n];
    const float v = fmaxf(sc + agg, 0.f) * mv;
    ftsOutF[((size_t)b * Wd + o) * N_ + n] = v;
    ftsOutH[((size_t)b * N_ + n) * Wd + o] = (_Float16)v;
  }
}

// ---------------------------------------------------------------------------
// Head: pooled -> fc(256) -> [cat global fc(64)] -> 128 -> 2. Tiny, VALU.
// ---------------------------------------------------------------------------
__global__ void head_kernel(const float* __restrict__ fts,
                            const float* __restrict__ counts,
                            const float* __restrict__ gfeat,
                            const float* __restrict__ Wfc, const float* __restrict__ bfc,
                            const float* __restrict__ Wg,  const float* __restrict__ bg,
                            const float* __restrict__ W1,  const float* __restrict__ b1,
                            const float* __restrict__ W2,  const float* __restrict__ b2,
                            float* __restrict__ out) {
  __shared__ float xv[256], h1[256], yv[64], z[128];
  const int b = blockIdx.x;
  const int t = threadIdx.x;               // 256 threads
  {
    const float* f = fts + ((size_t)b * 256 + t) * N_;
    float s = 0.f;
    for (int n = 0; n < N_; ++n) s += f[n];
    xv[t] = s / counts[b];
  }
  __syncthreads();
  {
    float s = bfc[t];
    const float* w = Wfc + (size_t)t * 256;
    for (int c = 0; c < 256; ++c) s = fmaf(w[c], xv[c], s);
    h1[t] = fmaxf(s, 0.f);
  }
  if (t < 64) {
    float s = bg[t];
    const float* w = Wg + (size_t)t * 4;
    for (int c = 0; c < 4; ++c) s = fmaf(w[c], gfeat[b * 4 + c], s);
    yv[t] = fmaxf(s, 0.f);
  }
  __syncthreads();
  if (t < 128) {
    float s = b1[t];
    const float* w = W1 + (size_t)t * 320;
    for (int c = 0; c < 256; ++c) s = fmaf(w[c], h1[c], s);
    for (int c = 0; c < 64; ++c)  s = fmaf(w[256 + c], yv[c], s);
    z[t] = fmaxf(s, 0.f);
  }
  __syncthreads();
  if (t < 2) {
    float s = b2[t];
    const float* w = W2 + (size_t)t * 128;
    for (int c = 0; c < 128; ++c) s = fmaf(w[c], z[c], s);
    out[b * 2 + t] = s;
  }
}

// ---------------------------------------------------------------------------
extern "C" void kernel_launch(void* const* d_in, const int* in_sizes, int n_in,
                              void* d_out, int out_size, void* d_ws, size_t ws_size,
                              hipStream_t stream) {
  (void)in_sizes; (void)n_in; (void)out_size; (void)ws_size;

  const float* points   = (const float*)d_in[0];
  const float* features = (const float*)d_in[1];
  const float* gfeat    = (const float*)d_in[2];
  const float* bn_g     = (const float*)d_in[3];
  const float* bn_b     = (const float*)d_in[4];

  // Workspace carve-up (256B aligned).
  char* ws = (char*)d_ws;
  size_t off = 0;
  auto take = [&](size_t bytes) -> char* {
    char* p = ws + off;
    off = (off + bytes + 255) & ~(size_t)255;
    return p;
  };
  float* mask    = (float*)take((size_t)B_ * N_ * 4);
  float* counts  = (float*)take((size_t)B_ * 4);
  int*   nbr     = (int*)take((size_t)B_ * N_ * K_ * 4);
  float* fts32   = (float*)take((size_t)B_ * 256 * N_ * 4);    // [b][c][n], knn/head
  _Float16* ftsHA = (_Float16*)take((size_t)B_ * N_ * 256 * 2); // [b][n][C] ping
  _Float16* ftsHB = (_Float16*)take((size_t)B_ * N_ * 256 * 2); // [b][n][C] pong
  _Float16* wbuf  = (_Float16*)take((size_t)640 * 1024);        // f16 weights (<600KB)

  const int CinArr[3]  = {7, 64, 128};
  const int CpadArr[3] = {32, 64, 128};
  const int WdArr[3]   = {64, 128, 256};

  _Float16 *w0p[3], *w1p[3], *w2p[3], *wscp[3];
  {
    _Float16* p = wbuf;
    for (int i = 0; i < 3; ++i) {
      const int Wd = WdArr[i], Cpad = CpadArr[i];
      w0p[i]  = p; p += (size_t)Wd * 2 * Cpad;   // == Wd*Wd (2*Cpad == Wd)
      w1p[i]  = p; p += (size_t)Wd * Wd;
      w2p[i]  = p; p += (size_t)Wd * Wd;
      wscp[i] = p; p += (size_t)Wd * Cpad;
    }
  }

  // 1) prep (writes f16 fts0 in [b][n][32] staging layout)
  prep_kernel<<<dim3(B_), dim3(N_), 0, stream>>>(features, bn_g, bn_b,
                                                 mask, counts, ftsHA);

  // 2) weight repack f32 -> f16
  for (int i = 0; i < 3; ++i) {
    const int Ei = 5 + 12 * i;
    const int Cin = CinArr[i], Cpad = CpadArr[i], Wd = WdArr[i];
    const int n0 = Wd * 2 * Cpad;
    conv_w_edge0<<<dim3((n0 + 255) / 256), dim3(256), 0, stream>>>(
        (const float*)d_in[Ei + 6], w0p[i], Cin, Cpad, Wd);
    conv_w_plain<<<dim3((Wd * Wd + 255) / 256), dim3(256), 0, stream>>>(
        (const float*)d_in[Ei + 7], w1p[i], Wd * Wd);
    conv_w_plain<<<dim3((Wd * Wd + 255) / 256), dim3(256), 0, stream>>>(
        (const float*)d_in[Ei + 8], w2p[i], Wd * Wd);
    conv_w_sc<<<dim3((Wd * Cpad + 255) / 256), dim3(256), 0, stream>>>(
        (const float*)d_in[Ei + 9], wscp[i], Cin, Cpad, Wd);
  }

  // 3) three EdgeConv blocks (knn + fused async-stage/gather/GEMM/aggregate)
  _Float16* hIn = ftsHA;
  _Float16* hOut = ftsHB;
  for (int i = 0; i < 3; ++i) {
    const int Ei = 5 + 12 * i;
    const int Cin = CinArr[i], Cpad = CpadArr[i], Wd = WdArr[i];

    // knn coords: block0 -> raw points (C=2); else f32 fts from previous block.
    const float* coords = (i == 0) ? points : fts32;
    const int Ck = (i == 0) ? 2 : Cin;
    const size_t knnSh = (size_t)Ck * N_ * 4 + (size_t)N_ * 4;
    knn_kernel<<<dim3(B_), dim3(N_), knnSh, stream>>>(coords, mask, Ck, nbr);

    const size_t edgeSh = (size_t)N_ * Cpad * 2 + 2 * (size_t)256 * Wd * 2;
    edge_block_kernel<<<dim3(B_ * 8), dim3(256), edgeSh, stream>>>(
        hIn, fts32, hOut, nbr, mask,
        w0p[i], w1p[i], w2p[i], wscp[i],
        (const float*)d_in[Ei + 0], (const float*)d_in[Ei + 1],
        (const float*)d_in[Ei + 2], (const float*)d_in[Ei + 3],
        (const float*)d_in[Ei + 4], (const float*)d_in[Ei + 5],
        (const float*)d_in[Ei + 10], (const float*)d_in[Ei + 11],
        Cpad, Wd);

    _Float16* tmp = hIn; hIn = hOut; hOut = tmp;
  }

  // 4) head (fts32 holds final 256-ch features from block 3)
  head_kernel<<<dim3(B_), dim3(256), 0, stream>>>(
      fts32, counts, gfeat,
      (const float*)d_in[41], (const float*)d_in[42],
      (const float*)d_in[47], (const float*)d_in[48],
      (const float*)d_in[43], (const float*)d_in[44],
      (const float*)d_in[45], (const float*)d_in[46],
      (float*)d_out);
}
